// GraphFNO_46093589021193
// MI455X (gfx1250) — matrix operand order repaired and checked by reading, not compiled, not run on hardware
//
#include <hip/hip_runtime.h>
#include <hip/hip_bf16.h>

// ---------------- problem constants (match reference) ----------------
#define Lc   4
#define Wd   256
#define NFc  256
#define Bc   2
#define Nn   16384
#define Ec   524288
#define Mall (Bc * Nn)     // 32768 rows

typedef __attribute__((ext_vector_type(16))) __bf16 v16bf;
typedef __attribute__((ext_vector_type(8)))  float  v8f;

__device__ __forceinline__ __bf16 f2bf(float x) { return (__bf16)x; }

// ---------------- async global->LDS (probe-verified on this toolchain) ----------------
#if defined(__HIP_DEVICE_COMPILE__) && \
    __has_builtin(__builtin_amdgcn_global_load_async_to_lds_b128) && \
    __has_builtin(__builtin_amdgcn_global_load_async_to_lds_b32) && \
    __has_builtin(__builtin_amdgcn_s_wait_asynccnt)
#define HAVE_ASYNC 1
#else
#define HAVE_ASYNC 0
#endif

typedef __attribute__((__vector_size__(4 * sizeof(int)))) int vi4;

__device__ __forceinline__ void cp4(const float* g, float* l)
{
#if HAVE_ASYNC
    __builtin_amdgcn_global_load_async_to_lds_b128((vi4*)g, (vi4*)l, 0, 0);
#else
    __builtin_prefetch(g + 32, 0, 0);
    *(float4*)l = *(const float4*)g;
#endif
}

__device__ __forceinline__ void cp1(const float* g, float* l)
{
#if HAVE_ASYNC
    __builtin_amdgcn_global_load_async_to_lds_b32((int*)g, (int*)l, 0, 0);
#else
    *l = *g;
#endif
}

__device__ __forceinline__ void tile_wait_sync()
{
#if HAVE_ASYNC
    __builtin_amdgcn_s_wait_asynccnt(0);
#endif
    __syncthreads();
}

// =====================================================================
// WMMA GEMM, LDS-tiled, double-buffered (async DMA), split-K capable.
//   out[M,N] = act( A @ B (+bias) (+addsrc) ),  A row-major [M,K] f32
//   BMODE 0: B is torch weight w[N,K] (K contiguous)  -> out = A @ w^T
//   BMODE 1: B is activation   b[K,N] (N contiguous)  -> out = A @ b
// Block tile 128x64x32, 8 waves, each wave computes 16x64 (4 wmma/k-step,
// issued back-to-back after preloading all fragments).
// gridDim.z > 1 => split-K, partials accumulated with atomicAdd (out pre-zeroed).
// =====================================================================
#define BKP 36   // padded LDS row stride (32 + 4)

template <int BMODE>
__global__ __launch_bounds__(256)
void gemm_wmma(const float* __restrict__ A0, const float* __restrict__ A1,
               int ksplit, int lda0, int lda1,
               const float* __restrict__ Bm, int ldb,
               const float* __restrict__ bias,
               const float* __restrict__ addsrc,
               float* __restrict__ out, int ldo,
               int Mr, int Nc, int K,
               long long sA, long long sB, long long sO, int act)
{
    __shared__ float ldsA[2][128 * BKP];
    __shared__ float ldsB[2][64 * BKP];

    const int tid  = threadIdx.x;
    const int lane = tid & 31;
    const int lo   = lane & 15;
    const int hi   = lane >> 4;
    const int wave = tid >> 5;

    const int tmn = Nc >> 6;                      // N tiles of 64
    const int tm  = blockIdx.x / tmn;
    const int tn  = blockIdx.x - tm * tmn;
    const int Mb  = tm << 7;                      // 128-row block
    const int Nb  = tn << 6;                      // 64-col block

    const long long b = blockIdx.y;
    A0 += b * sA;  A1 += b * sA;  Bm += b * sB;  out += b * sO;
    const float* asrc = addsrc ? addsrc + b * sO : nullptr;

    const int kchunks = gridDim.z;
    const int klen = K / kchunks;
    const int kbeg = blockIdx.z * klen;
    const int nk   = klen >> 5;                   // k-steps of 32

    // staging thread mapping
    const int arow = tid >> 1;                    // 0..127
    const int acg  = (tid & 1) << 4;              // 0 or 16
    const int bn   = tid & 63;                    // 0..63
    const int bkg  = (tid >> 6) << 3;             // 0,8,16,24

    auto stage = [&](int bufi, int k0) {
        // ---- A tile: 128x32, row-major, K contiguous (split-A aware) ----
        const float* Ar; int kk;
        if (k0 < ksplit) { Ar = A0 + (size_t)(Mb + arow) * lda0; kk = k0; }
        else             { Ar = A1 + (size_t)(Mb + arow) * lda1; kk = k0 - ksplit; }
        const float* ga = Ar + kk + acg;
        float* la = &ldsA[bufi][arow * BKP + acg];
        cp4(ga + 0, la + 0);  cp4(ga + 4,  la + 4);
        cp4(ga + 8, la + 8);  cp4(ga + 12, la + 12);
        // ---- B tile stored [n][k] in LDS ----
        float* lb = &ldsB[bufi][bn * BKP + bkg];
        if (BMODE == 0) {
            const float* gb = Bm + (size_t)(Nb + bn) * ldb + k0 + bkg;
            cp4(gb, lb);  cp4(gb + 4, lb + 4);
        } else {
#pragma unroll
            for (int j = 0; j < 8; ++j)           // transpose via per-lane b32 DMA
                cp1(Bm + (size_t)(k0 + bkg + j) * ldb + Nb + bn, lb + j);
        }
    };

    v8f acc[4] = {};
    stage(0, kbeg);
    for (int kt = 0; kt < nk; ++kt) {
        tile_wait_sync();
        if (kt + 1 < nk) stage((kt + 1) & 1, kbeg + ((kt + 1) << 5));

        // preload A fragment
        const float* pa = &ldsA[kt & 1][(wave * 16 + lo) * BKP];
        v16bf a;
#pragma unroll
        for (int e = 0; e < 8; ++e) a[e]     = f2bf(pa[(hi << 3) + e]);        // K 0..7/8..15
#pragma unroll
        for (int e = 0; e < 8; ++e) a[8 + e] = f2bf(pa[16 + (hi << 3) + e]);   // K 16..31
        // preload all four B fragments (independent registers -> batched ds loads)
        v16bf bf[4];
#pragma unroll
        for (int nt = 0; nt < 4; ++nt) {
            const float* pb = &ldsB[kt & 1][(nt * 16 + lo) * BKP + (hi << 3)];
#pragma unroll
            for (int e = 0; e < 16; ++e) bf[nt][e] = f2bf(pb[(e < 8) ? e : e + 8]); // K 0..7/8..15, 16..31
        }
        // back-to-back matrix ops
#pragma unroll
        for (int nt = 0; nt < 4; ++nt)
            acc[nt] = __builtin_amdgcn_wmma_f32_16x16x32_bf16(
                          false, a, false, bf[nt], (short)0, acc[nt], false, false);
    }

    // ---- epilogue: C rows hi*8+r, col lo ----
    const bool partial = (kchunks > 1);
#pragma unroll
    for (int nt = 0; nt < 4; ++nt) {
        int col = Nb + nt * 16 + lo;
#pragma unroll
        for (int r = 0; r < 8; ++r) {
            int row = Mb + wave * 16 + (hi << 3) + r;
            float v = acc[nt][r];
            if (partial) {
                atomicAdd(&out[(size_t)row * ldo + col], v);
            } else {
                if (bias) v += bias[col];
                if (asrc) v += asrc[(size_t)row * ldo + col];
                if (act == 1)      v = fmaxf(v, 0.f);
                else if (act == 2) v = 0.5f * v * (1.f + erff(v * 0.70710678118f));
                out[(size_t)row * ldo + col] = v;
            }
        }
    }
}

// ---------------- h = x @ p_w^T + p_b  (K=3, trivial) ----------------
__global__ void k_embed(const float* __restrict__ x, const float* __restrict__ pw,
                        const float* __restrict__ pb, float* __restrict__ h)
{
    int i = blockIdx.x * blockDim.x + threadIdx.x;
    if (i >= Mall * Wd) return;
    int m = i >> 8, w = i & 255;
    h[i] = pb[w] + x[m*3+0]*pw[w*3+0] + x[m*3+1]*pw[w*3+1] + x[m*3+2]*pw[w*3+2];
}

// ------------- Ut[b][f][n] = U[b][n][f], 32x32 LDS-tiled (coalesced both sides) -------------
__global__ void k_transpose_U(const float* __restrict__ U, float* __restrict__ Ut)
{
    __shared__ float tile[32][33];
    const int n0 = blockIdx.x << 5;               // node tile
    const int f0 = blockIdx.y << 5;               // freq tile
    const size_t base = (size_t)blockIdx.z * Nn * NFc;
    const int tf = threadIdx.x & 31;
    const int tr = threadIdx.x >> 5;              // 0..7
#pragma unroll
    for (int j = 0; j < 32; j += 8)               // load: coalesced along f
        tile[tr + j][tf] = U[base + (size_t)(n0 + tr + j) * NFc + f0 + tf];
    __syncthreads();
#pragma unroll
    for (int j = 0; j < 32; j += 8)               // store: coalesced along n
        Ut[base + (size_t)(f0 + tr + j) * Nn + n0 + tf] = tile[tf][tr + j];
}

// -------- out_ft[b][f][o] += sum_{i in chunk} xft[b][f][i] * four_w[i][o][f] --------
// i split into 4 chunks (grid.y) for occupancy; oft pre-zeroed, f32 atomics.
__global__ void k_mix(const float* __restrict__ xft, const float* __restrict__ fw,
                      float* __restrict__ oft)
{
    int o  = blockIdx.x;         // 0..255
    int i0 = blockIdx.y << 6;    // i-chunk of 64
    int f  = threadIdx.x;        // 0..255, coalesced over four_w's last dim
    float a0 = 0.f, a1 = 0.f;
    const float* w  = fw + (size_t)o * NFc + f;
    const float* x0 = xft + (size_t)f * Wd;
    const float* x1 = x0 + (size_t)NFc * Wd;
    for (int i = i0; i < i0 + 64; ++i) {
        float wv = w[(size_t)i * Wd * NFc];
        a0 += x0[i] * wv;
        a1 += x1[i] * wv;
    }
    atomicAdd(&oft[(size_t)f * Wd + o], a0);
    atomicAdd(&oft[(size_t)NFc * Wd + (size_t)f * Wd + o], a1);
}

// ---------------- in-place layernorm over last dim (256) ----------------
__global__ void k_layernorm(float* __restrict__ t, const float* __restrict__ s,
                            const float* __restrict__ bb)
{
    __shared__ float s1[256], s2[256];
    int row = blockIdx.x, tid = threadIdx.x;
    float v = t[(size_t)row * Wd + tid];
    s1[tid] = v; s2[tid] = v * v; __syncthreads();
    for (int off = 128; off; off >>= 1) {
        if (tid < off) { s1[tid] += s1[tid + off]; s2[tid] += s2[tid + off]; }
        __syncthreads();
    }
    float mean = s1[0] * (1.f / Wd);
    float var  = s2[0] * (1.f / Wd) - mean * mean;
    t[(size_t)row * Wd + tid] = (v - mean) * rsqrtf(var + 1e-5f) * s[tid] + bb[tid];
}

// ---------------- per-edge gate MLP ----------------
__global__ void k_gate(const float* __restrict__ ewt, const int* __restrict__ ei,
                       const float* __restrict__ lif,
                       const float* __restrict__ eww, const float* __restrict__ ewb,
                       const float* __restrict__ g1w, const float* __restrict__ g1b,
                       const float* __restrict__ g2w, const float* __restrict__ g2b,
                       float* __restrict__ g)
{
    int e = blockIdx.x * blockDim.x + threadIdx.x;
    if (e >= Ec) return;
    int s = ei[e], d = ei[Ec + e];
    float wv = ewt[e];
    float gin[48];
#pragma unroll
    for (int i = 0; i < 16; ++i) gin[i]      = wv * eww[i] + ewb[i];
#pragma unroll
    for (int i = 0; i < 16; ++i) gin[16 + i] = lif[(size_t)d * 16 + i];
#pragma unroll
    for (int i = 0; i < 16; ++i) gin[32 + i] = lif[(size_t)s * 16 + i];
    float h3[3];
#pragma unroll
    for (int j = 0; j < 3; ++j) {
        float a = g1b[j];
#pragma unroll
        for (int k = 0; k < 48; ++k) a += g1w[j * 48 + k] * gin[k];
        h3[j] = fmaxf(a, 0.f);
    }
    float z = g2b[0] + g2w[0] * h3[0] + g2w[1] * h3[1] + g2w[2] * h3[2];
    g[e] = 1.f / (1.f + expf(-z));
}

__global__ void k_fill0(float* __restrict__ p, int total)
{
    int i = blockIdx.x * blockDim.x + threadIdx.x;
    if (i < total) p[i] = 0.f;
}

// ---------------- msg scatter: agg[dst] += xs[src] * g[e] ----------------
__global__ void k_scatter(const float* __restrict__ xs, const float* __restrict__ g,
                          const int* __restrict__ ei, float* __restrict__ agg)
{
    int e = blockIdx.x;
    int w = threadIdx.x;
    int s = ei[e], d = ei[Ec + e];
    float val = xs[(size_t)s * Wd + w] * g[e];
    atomicAdd(&agg[(size_t)d * Wd + w], val);
}

// ---------------- xs = agg / max(||agg||_2, 1e-12) per row ----------------
__global__ void k_l2norm(const float* __restrict__ agg, float* __restrict__ xs)
{
    __shared__ float s1[256];
    int row = blockIdx.x, tid = threadIdx.x;
    float v = agg[(size_t)row * Wd + tid];
    s1[tid] = v * v; __syncthreads();
    for (int off = 128; off; off >>= 1) {
        if (tid < off) s1[tid] += s1[tid + off];
        __syncthreads();
    }
    float nrm = fmaxf(sqrtf(s1[0]), 1e-12f);
    xs[(size_t)row * Wd + tid] = v / nrm;
}

// ---------------- out[m] = q1[m,:] . q2_w + q2_b  (one wave per row) ----------------
__global__ void k_q2(const float* __restrict__ q1, const float* __restrict__ w,
                     const float* __restrict__ bb, float* __restrict__ out)
{
    int tid = threadIdx.x;
    int wid = tid >> 5, lane = tid & 31;
    int row = blockIdx.x * 8 + wid;
    float acc = 0.f;
#pragma unroll
    for (int t = 0; t < 4; ++t)
        acc += q1[(size_t)row * 128 + lane + 32 * t] * w[lane + 32 * t];
#pragma unroll
    for (int m = 16; m; m >>= 1) acc += __shfl_xor(acc, m, 32);
    if (lane == 0) out[row] = acc + bb[0];
}

// =====================================================================
extern "C" void kernel_launch(void* const* d_in, const int* in_sizes, int n_in,
                              void* d_out, int out_size, void* d_ws, size_t ws_size,
                              hipStream_t stream)
{
    (void)in_sizes; (void)n_in; (void)out_size; (void)ws_size;
    const float* x     = (const float*)d_in[0];
    const float* U     = (const float*)d_in[1];
    const int*   ei    = (const int*)  d_in[2];
    const float* ewt   = (const float*)d_in[3];
    const float* lif   = (const float*)d_in[4];
    const float* p_w   = (const float*)d_in[5];
    const float* p_b   = (const float*)d_in[6];
    const float* fourw = (const float*)d_in[7];
    const float* w_w   = (const float*)d_in[8];
    const float* w_b   = (const float*)d_in[9];
    const float* ln_s  = (const float*)d_in[10];
    const float* ln_b  = (const float*)d_in[11];
    const float* lin_w = (const float*)d_in[12];
    const float* lin_b = (const float*)d_in[13];
    const float* ew_w  = (const float*)d_in[14];
    const float* ew_b  = (const float*)d_in[15];
    const float* g1_w  = (const float*)d_in[16];
    const float* g1_b  = (const float*)d_in[17];
    const float* g2_w  = (const float*)d_in[18];
    const float* g2_b  = (const float*)d_in[19];
    const float* projw = (const float*)d_in[20];
    const float* projb = (const float*)d_in[21];
    const float* q1_w  = (const float*)d_in[22];
    const float* q1_b  = (const float*)d_in[23];
    const float* q2_w  = (const float*)d_in[24];
    const float* q2_b  = (const float*)d_in[25];
    float* outp = (float*)d_out;

    // ---------------- workspace bump allocator ----------------
    char* wsp = (char*)d_ws;
    auto alloc = [&](size_t bytes) -> float* {
        float* p = (float*)wsp;
        wsp += (bytes + 255) & ~(size_t)255;
        return p;
    };
    float* h    = alloc((size_t)Mall * Wd * 4);
    float* Ut   = alloc((size_t)Bc * NFc * Nn * 4);
    float* xft  = alloc((size_t)Bc * NFc * Wd * 4);
    float* oft  = alloc((size_t)Bc * NFc * Wd * 4);
    float* t1   = alloc((size_t)Mall * Wd * 4);
    float* xs   = alloc((size_t)Mall * Wd * 4);
    float* agg  = alloc((size_t)Mall * Wd * 4);
    float* gbuf = alloc((size_t)Ec * 4);
    float* q1b  = alloc((size_t)Mall * 128 * 4);

    // ---------------- GEMM launchers (block tile 128x64) ----------------
    auto launchBT = [&](const float* A0, const float* A1, int ksplit, int lda0, int lda1,
                        const float* Bm, int ldb, const float* bias, const float* addsrc,
                        float* out, int ldo, int Mr, int Nc, int K, int act) {
        dim3 grid((Mr / 128) * (Nc / 64), 1, 1);
        gemm_wmma<0><<<grid, 256, 0, stream>>>(A0, A1, ksplit, lda0, lda1, Bm, ldb,
                                               bias, addsrc, out, ldo, Mr, Nc, K,
                                               0, 0, 0, act);
    };
    auto launchBN = [&](const float* A0, int lda0,
                        const float* Bm, int ldb, float* out, int ldo,
                        int Mr, int Nc, int K,
                        long long sA, long long sB, long long sO,
                        int batch, int kchunks) {
        dim3 grid((Mr / 128) * (Nc / 64), batch, kchunks);
        gemm_wmma<1><<<grid, 256, 0, stream>>>(A0, A0, K, lda0, lda0, Bm, ldb,
                                               nullptr, nullptr, out, ldo, Mr, Nc, K,
                                               sA, sB, sO, 0);
    };

    // ---------------- prologue ----------------
    k_embed<<<(Mall * Wd + 255) / 256, 256, 0, stream>>>(x, p_w, p_b, h);
    {
        dim3 tgrid(Nn / 32, NFc / 32, Bc);
        k_transpose_U<<<tgrid, 256, 0, stream>>>(U, Ut);
    }

    // ---------------- layers ----------------
    for (int l = 0; l < Lc; ++l) {
        const float* fwl  = fourw + (size_t)l * Wd * Wd * NFc;
        const float* wwl  = w_w   + (size_t)l * Wd * Wd;
        const float* wbl  = w_b   + (size_t)l * Wd;
        const float* lsl  = ln_s  + (size_t)l * Wd;
        const float* lbl  = ln_b  + (size_t)l * Wd;
        const float* lwl  = lin_w + (size_t)l * Wd * Wd;
        const float* lbb  = lin_b + (size_t)l * Wd;
        const float* ewwl = ew_w  + (size_t)l * 16;
        const float* ewbl = ew_b  + (size_t)l * 16;
        const float* g1wl = g1_w  + (size_t)l * 144;
        const float* g1bl = g1_b  + (size_t)l * 3;
        const float* g2wl = g2_w  + (size_t)l * 3;
        const float* g2bl = g2_b  + (size_t)l * 1;

        // x_ft[b] = Ut[b] @ h[b]   [256x256] += over 32 K-chunks of 512
        k_fill0<<<(Bc * NFc * Wd + 255) / 256, 256, 0, stream>>>(xft, Bc * NFc * Wd);
        launchBN(Ut, Nn, h, Wd, xft, Wd, NFc, Wd, Nn,
                 (long long)NFc * Nn, (long long)Nn * Wd, (long long)NFc * Wd,
                 Bc, 32);

        // out_ft: per-frequency channel mix (streams four_w once, 4 i-chunks)
        k_fill0<<<(Bc * NFc * Wd + 255) / 256, 256, 0, stream>>>(oft, Bc * NFc * Wd);
        {
            dim3 mgrid(Wd, 4);
            k_mix<<<mgrid, NFc, 0, stream>>>(xft, fwl, oft);
        }

        // t1[b] = U[b] @ out_ft[b]   [16384x256] = [16384x256] @ [256x256]
        launchBN(U, NFc, oft, Wd, t1, Wd, Nn, Wd, NFc,
                 (long long)Nn * NFc, (long long)NFc * Wd, (long long)Nn * Wd,
                 Bc, 1);

        // xf = gelu(t1 + h @ w_w^T + w_b)   (fused epilogue, in place)
        launchBT(h, h, Wd, Wd, Wd, wwl, Wd, wbl, t1, t1, Wd, Mall, Wd, Wd, 2);
        k_layernorm<<<Mall, Wd, 0, stream>>>(t1, lsl, lbl);

        // graph branch: xs_pre = h @ lin_w^T + lin_b
        launchBT(h, h, Wd, Wd, Wd, lwl, Wd, lbb, nullptr, xs, Wd, Mall, Wd, Wd, 0);
        k_gate<<<(Ec + 255) / 256, 256, 0, stream>>>(ewt, ei, lif, ewwl, ewbl,
                                                     g1wl, g1bl, g2wl, g2bl, gbuf);
        k_fill0<<<(Mall * Wd + 255) / 256, 256, 0, stream>>>(agg, Mall * Wd);
        k_scatter<<<Ec, Wd, 0, stream>>>(xs, gbuf, ei, agg);
        k_l2norm<<<Mall, Wd, 0, stream>>>(agg, xs);

        // h = [xf | xs] @ proj_w^T + proj_b   (split-A GEMM, K = 512)
        launchBT(t1, xs, Wd, Wd, Wd, projw, 2 * Wd, projb, nullptr, h, Wd,
                 Mall, Wd, 2 * Wd, 0);
    }

    // ---------------- head: q MLP ----------------
    launchBT(h, h, Wd, Wd, Wd, q1_w, Wd, q1_b, nullptr, q1b, 128,
             Mall, 128, Wd, 1);
    k_q2<<<Mall / 8, 256, 0, stream>>>(q1b, q2_w, q2_b, outp);
}
